// SelfAttention_17592186044608
// MI455X (gfx1250) — matrix operand order repaired and checked by reading, not compiled, not run on hardware
//
#include <hip/hip_runtime.h>
#include <hip/hip_bf16.h>

typedef __attribute__((ext_vector_type(16))) __bf16       v16bf;
typedef __attribute__((ext_vector_type(8)))  float        v8f;
typedef __attribute__((ext_vector_type(4)))  __bf16       bf16x4;
typedef __attribute__((ext_vector_type(4)))  unsigned int u32x4;
typedef int v4i __attribute__((vector_size(16)));

// Gate the CDNA5 async global->LDS path on builtin availability so the file
// always compiles; fallback is the sync VGPR-staging path.
#if defined(__has_builtin)
#if __has_builtin(__builtin_amdgcn_global_load_async_to_lds_b128) && \
    __has_builtin(__builtin_amdgcn_s_wait_asynccnt)
#define GEMM_ASYNC 1
#endif
#endif
#ifndef GEMM_ASYNC
#define GEMM_ASYNC 0
#endif

#if GEMM_ASYNC
__device__ __forceinline__ void async_ld16(const __bf16* g, __bf16* s) {
  // Per-lane 16B copy: LDS[s] = MEM[g]; tracked by ASYNCcnt.
  // Builtin signature (from hipcc diagnostic): (v4i AS1*, v4i AS3*, imm, imm).
  __builtin_amdgcn_global_load_async_to_lds_b128(
      (__attribute__((address_space(1))) v4i*)(unsigned long long)(uintptr_t)g,
      (__attribute__((address_space(3))) v4i*)(unsigned)(uintptr_t)s,
      0, 0);
}
#endif

// ---------------------------------------------------------------------------
// Elementwise f32 -> bf16 conversion, 4 elements / thread
// ---------------------------------------------------------------------------
__global__ __launch_bounds__(256)
void cvt_f32_bf16_x4(const float* __restrict__ in, __bf16* __restrict__ out, long n4) {
  long i = (long)blockIdx.x * 256 + threadIdx.x;
  if (i >= n4) return;
  float4 f = ((const float4*)in)[i];
  bf16x4 o;
  o.x = (__bf16)f.x; o.y = (__bf16)f.y; o.z = (__bf16)f.z; o.w = (__bf16)f.w;
  ((bf16x4*)out)[i] = o;
}

// ---------------------------------------------------------------------------
// Transpose + convert: in f32 [K][N] row-major -> out bf16 [N][K] row-major
// ---------------------------------------------------------------------------
__global__ __launch_bounds__(256)
void transpose_cvt(const float* __restrict__ in, __bf16* __restrict__ out,
                   int K, int N) {
  __shared__ float tile[32][33];
  int k0 = blockIdx.y * 32, n0 = blockIdx.x * 32;
  int tx = threadIdx.x, ty = threadIdx.y;
  #pragma unroll
  for (int i = ty; i < 32; i += 8)
    tile[i][tx] = in[(long)(k0 + i) * N + (n0 + tx)];
  __syncthreads();
  #pragma unroll
  for (int i = ty; i < 32; i += 8)
    out[(long)(n0 + i) * K + (k0 + tx)] = (__bf16)tile[tx][i];
}

// ---------------------------------------------------------------------------
// NT GEMM: D[M,N] = A[M,K] * B[N,K]^T (+ epilogue), bf16 in, f32 acc.
// 128x128 tile / workgroup (8 waves); wave -> 32x64 via 2x4 grid of
// v_wmma_f32_16x16x32_bf16; K-step 32; double-buffered LDS fed by
// global_load_async_to_lds_b128 (ASYNCcnt pipeline) when available.
// MODE 0: bf16 out +bias[n].  MODE 1: f32 out.  MODE 2: f32 out = resid+g*acc.
// ---------------------------------------------------------------------------
#define LSTR 40   // 32 K-elements + 8 pad (80B row stride, 16B-aligned chunks)

template<int MODE>
__global__ __launch_bounds__(256)
void gemm_nt(const __bf16* __restrict__ A, long sA, int lda,
             const __bf16* __restrict__ Bm, long sB, int ldb,
             void* __restrict__ Dv, long sD, int ldd,
             const float* __restrict__ bias,
             const float* __restrict__ resid,
             const float* __restrict__ gamma,
             int K)
{
  __shared__ __align__(16) __bf16 As[2][128 * LSTR];
  __shared__ __align__(16) __bf16 Bs[2][128 * LSTR];

  const int tid   = threadIdx.x;
  const int wave  = tid >> 5;
  const int lane  = tid & 31;
  const int hh    = lane >> 4;   // lane half (0: lanes 0-15, 1: lanes 16-31)
  const int lrow  = lane & 15;
  const int waveM = wave >> 1;   // 0..3  -> 32-row strip
  const int waveN = wave & 1;    // 0..1  -> 64-col strip

  const __bf16* Ab = A  + (long)blockIdx.z * sA + (long)blockIdx.y * 128 * lda;
  const __bf16* Bb = Bm + (long)blockIdx.z * sB + (long)blockIdx.x * 128 * ldb;

  // cooperative tile-load mapping: thread -> (row, 16-elem column half)
  const int lr = tid >> 1;
  const int lc = (tid & 1) * 16;
  const __bf16* ga = Ab + (long)lr * lda + lc;   // + k0 per step
  const __bf16* gb = Bb + (long)lr * ldb + lc;

  v8f acc[2][4] = {};

  // Shared per-tile WMMA body: fragments from one LDS buffer, 8 WMMAs.
  auto compute_tile = [&](const __bf16* Asb, const __bf16* Bsb) {
    v16bf afrag[2];
    #pragma unroll
    for (int fm = 0; fm < 2; fm++) {
      int ar = waveM * 32 + fm * 16 + lrow;
      u32x4 lo = *(const u32x4*)(Asb + ar * LSTR + hh * 8);
      u32x4 hi = *(const u32x4*)(Asb + ar * LSTR + 16 + hh * 8);
      afrag[fm] = __builtin_bit_cast(
          v16bf, __builtin_shufflevector(lo, hi, 0, 1, 2, 3, 4, 5, 6, 7));
    }
    v16bf bfrag[4];
    #pragma unroll
    for (int fn = 0; fn < 4; fn++) {
      int br = waveN * 64 + fn * 16 + lrow;
      u32x4 lo = *(const u32x4*)(Bsb + br * LSTR + hh * 8);
      u32x4 hi = *(const u32x4*)(Bsb + br * LSTR + 16 + hh * 8);
      bfrag[fn] = __builtin_bit_cast(
          v16bf, __builtin_shufflevector(lo, hi, 0, 1, 2, 3, 4, 5, 6, 7));
    }
    #pragma unroll
    for (int fm = 0; fm < 2; fm++)
      #pragma unroll
      for (int fn = 0; fn < 4; fn++)
        acc[fm][fn] = __builtin_amdgcn_wmma_f32_16x16x32_bf16(
            false, afrag[fm], false, bfrag[fn], (short)0, acc[fm][fn],
            false, false);
  };

#if GEMM_ASYNC
  // ---- ASYNCcnt-pipelined path: prefetch tile i+1 while computing tile i.
  auto prefetch = [&](int k0, int buf) {
    __bf16* sa  = &As[buf][lr * LSTR + lc];
    __bf16* sb2 = &Bs[buf][lr * LSTR + lc];
    async_ld16(ga + k0,     sa);
    async_ld16(ga + k0 + 8, sa + 8);
    async_ld16(gb + k0,     sb2);
    async_ld16(gb + k0 + 8, sb2 + 8);
  };
  const int nsteps = K >> 5;
  prefetch(0, 0);
  for (int i = 0; i < nsteps; i++) {
    const int cur = i & 1;
    __builtin_amdgcn_s_wait_asynccnt(0);   // this wave's tile-i loads landed
    __syncthreads();                       // all waves' tile i in LDS; all waves
                                           // done reading buf cur^1 (tile i-1)
    if (i + 1 < nsteps) prefetch((i + 1) * 32, cur ^ 1);
    compute_tile(As[cur], Bs[cur]);
  }
#else
  // ---- Fallback: sync VGPR staging.
  for (int k0 = 0; k0 < K; k0 += 32) {
    u32x4 a0 = *(const u32x4*)(ga + k0);
    u32x4 a1 = *(const u32x4*)(ga + k0 + 8);
    u32x4 b0 = *(const u32x4*)(gb + k0);
    u32x4 b1 = *(const u32x4*)(gb + k0 + 8);
    __syncthreads();
    *(u32x4*)(&As[0][lr * LSTR + lc])     = a0;
    *(u32x4*)(&As[0][lr * LSTR + lc + 8]) = a1;
    *(u32x4*)(&Bs[0][lr * LSTR + lc])     = b0;
    *(u32x4*)(&Bs[0][lr * LSTR + lc + 8]) = b1;
    __syncthreads();
    compute_tile(As[0], Bs[0]);
  }
#endif

  const float g = (MODE == 2) ? gamma[0] : 0.0f;
  // C/D layout: VGPR v, lane(hh,lrow) -> D[hh*8 + v][lrow]
  #pragma unroll
  for (int fm = 0; fm < 2; fm++) {
    #pragma unroll
    for (int fn = 0; fn < 4; fn++) {
      int rowb = blockIdx.y * 128 + waveM * 32 + fm * 16 + hh * 8;
      int col  = blockIdx.x * 128 + waveN * 64 + fn * 16 + lrow;
      #pragma unroll
      for (int v = 0; v < 8; v++) {
        float val = acc[fm][fn][v];
        long  idx = (long)(rowb + v) * ldd + col;
        if (MODE == 0) {
          (((__bf16*)Dv) + (long)blockIdx.z * sD)[idx] = (__bf16)(val + bias[col]);
        } else if (MODE == 1) {
          (((float*)Dv) + (long)blockIdx.z * sD)[idx] = val;
        } else {
          const float* rb = resid + (long)blockIdx.z * sD;
          (((float*)Dv) + (long)blockIdx.z * sD)[idx] = rb[idx] + g * val;
        }
      }
    }
  }
}

// ---------------------------------------------------------------------------
// Row softmax over 256 f32 elements -> bf16 (one block per row)
// ---------------------------------------------------------------------------
__global__ __launch_bounds__(256)
void softmax256(const float* __restrict__ S, __bf16* __restrict__ out) {
  __shared__ float red[256];
  long row = blockIdx.x;
  int  t   = threadIdx.x;
  float v  = S[row * 256 + t];
  red[t] = v;
  __syncthreads();
  #pragma unroll
  for (int s = 128; s > 0; s >>= 1) {
    if (t < s) red[t] = fmaxf(red[t], red[t + s]);
    __syncthreads();
  }
  float m = red[0];
  __syncthreads();
  float e = __expf(v - m);
  red[t] = e;
  __syncthreads();
  #pragma unroll
  for (int s = 128; s > 0; s >>= 1) {
    if (t < s) red[t] += red[t + s];
    __syncthreads();
  }
  float sum = red[0];
  out[row * 256 + t] = (__bf16)(e / sum);
}

// ---------------------------------------------------------------------------
// Host-side orchestration
// ---------------------------------------------------------------------------
extern "C" void kernel_launch(void* const* d_in, const int* in_sizes, int n_in,
                              void* d_out, int out_size, void* d_ws, size_t ws_size,
                              hipStream_t stream) {
  (void)in_sizes; (void)n_in; (void)out_size; (void)ws_size;
  const float* x     = (const float*)d_in[0];
  const float* wq    = (const float*)d_in[1];
  const float* bq    = (const float*)d_in[2];
  const float* wk    = (const float*)d_in[3];
  const float* bk    = (const float*)d_in[4];
  const float* wv    = (const float*)d_in[5];
  const float* bv    = (const float*)d_in[6];
  const float* gamma = (const float*)d_in[7];

  constexpr int  Bn = 32, HW = 256, C = 2048, C8 = 256;
  constexpr long XEL = (long)HW * C;       // 524288 per batch
  constexpr long QEL = (long)HW * C8;      // 65536 per batch

  char*  ws  = (char*)d_ws;
  size_t off = 0;
  auto carve = [&](size_t bytes) -> void* {
    void* p = ws + off;
    off += (bytes + 255) & ~(size_t)255;
    return p;
  };
  __bf16* xb  = (__bf16*)carve((size_t)Bn * XEL * 2);   // x as bf16  [B][256][2048]
  __bf16* wqT = (__bf16*)carve((size_t)C8 * C * 2);     // Wq^T       [256][2048]
  __bf16* wkT = (__bf16*)carve((size_t)C8 * C * 2);
  __bf16* wvT = (__bf16*)carve((size_t)C * C * 2);      // Wv^T       [2048][2048]
  __bf16* qb  = (__bf16*)carve((size_t)Bn * QEL * 2);   // Q          [B][256][256]
  __bf16* kb  = (__bf16*)carve((size_t)Bn * QEL * 2);
  __bf16* vb  = (__bf16*)carve((size_t)Bn * XEL * 2);   // V          [B][256][2048]
  float*  sb  = (float*) carve((size_t)Bn * QEL * 4);   // scores f32 [B][256][256]
  __bf16* ab  = (__bf16*)carve((size_t)Bn * QEL * 2);   // softmax    [B][256][256]

  // 1) x -> bf16
  long n4 = (long)Bn * XEL / 4;
  cvt_f32_bf16_x4<<<dim3((unsigned)(n4 / 256)), dim3(256), 0, stream>>>(x, xb, n4);

  // 2) weight transposes (f32 [K][N] -> bf16 [N][K])
  transpose_cvt<<<dim3(C8 / 32, C / 32), dim3(32, 8), 0, stream>>>(wq, wqT, C, C8);
  transpose_cvt<<<dim3(C8 / 32, C / 32), dim3(32, 8), 0, stream>>>(wk, wkT, C, C8);
  transpose_cvt<<<dim3(C  / 32, C / 32), dim3(32, 8), 0, stream>>>(wv, wvT, C, C);

  // 3) Q = X*Wq + bq, K = X*Wk + bk  (M=256, N=256, K=2048)
  gemm_nt<0><<<dim3(C8 / 128, HW / 128, Bn), 256, 0, stream>>>(
      xb, XEL, C, wqT, 0L, C, qb, QEL, C8, bq, nullptr, nullptr, C);
  gemm_nt<0><<<dim3(C8 / 128, HW / 128, Bn), 256, 0, stream>>>(
      xb, XEL, C, wkT, 0L, C, kb, QEL, C8, bk, nullptr, nullptr, C);

  // 4) V = X*Wv + bv  (M=256, N=2048, K=2048)
  gemm_nt<0><<<dim3(C / 128, HW / 128, Bn), 256, 0, stream>>>(
      xb, XEL, C, wvT, 0L, C, vb, XEL, C, bv, nullptr, nullptr, C);

  // 5) S = Q*K^T (f32 scores; M=N=K=256)
  gemm_nt<1><<<dim3(C8 / 128, HW / 128, Bn), 256, 0, stream>>>(
      qb, QEL, C8, kb, QEL, C8, sb, QEL, C8, nullptr, nullptr, nullptr, C8);

  // 6) attn = softmax(S) rows -> bf16
  softmax256<<<dim3(Bn * HW), dim3(256), 0, stream>>>(sb, ab);

  // 7) out = x + gamma * (attn @ Vr^T), Vr = V buffer viewed [2048][256]
  //    (M=256, N=2048, K=256; batch-strided over the same vb buffer)
  gemm_nt<2><<<dim3(C / 128, HW / 128, Bn), 256, 0, stream>>>(
      ab, QEL, C8, vb, XEL, C8, (float*)d_out, XEL, C, nullptr, x, gamma, C8);
}
